// FP4Linear_9414568313144
// MI455X (gfx1250) — compile-verified, matrix operand order checked by reading, not run on hardware
//
#include <hip/hip_runtime.h>

typedef _Float16     v16h __attribute__((ext_vector_type(16)));
typedef _Float16     v8h  __attribute__((ext_vector_type(8)));
typedef float        v8f  __attribute__((ext_vector_type(8)));
typedef unsigned int u32x4 __attribute__((ext_vector_type(4)));
typedef int          i32x8 __attribute__((ext_vector_type(8)));
typedef int          i32x4 __attribute__((ext_vector_type(4)));

#define BM 128
#define BN 128
#define BK 32
#define THREADS 256

// ---------------- conversion kernels ----------------
__global__ __launch_bounds__(256)
void cvt_f32_to_f16(const float* __restrict__ in, _Float16* __restrict__ out) {
    size_t i = ((size_t)blockIdx.x * blockDim.x + threadIdx.x) * 8;
    const float4* p = (const float4*)(in + i);
    float4 a = p[0], b = p[1];
    v8h o;
    o[0] = (_Float16)a.x; o[1] = (_Float16)a.y;
    o[2] = (_Float16)a.z; o[3] = (_Float16)a.w;
    o[4] = (_Float16)b.x; o[5] = (_Float16)b.y;
    o[6] = (_Float16)b.z; o[7] = (_Float16)b.w;
    *(v8h*)(out + i) = o;
}

__global__ __launch_bounds__(256)
void cvt_i32_to_f16(const int* __restrict__ in, _Float16* __restrict__ out) {
    size_t i = ((size_t)blockIdx.x * blockDim.x + threadIdx.x) * 8;
    const int4* p = (const int4*)(in + i);
    int4 a = p[0], b = p[1];
    v8h o;
    o[0] = (_Float16)(float)a.x; o[1] = (_Float16)(float)a.y;
    o[2] = (_Float16)(float)a.z; o[3] = (_Float16)(float)a.w;
    o[4] = (_Float16)(float)b.x; o[5] = (_Float16)(float)b.y;
    o[6] = (_Float16)(float)b.z; o[7] = (_Float16)(float)b.w;
    *(v8h*)(out + i) = o;
}

// ---------------- TDM: 2D tile load (128 rows x 32 f16) into LDS ----------------
// D# per CDNA5 ISA ch.8: group0 = {flags, lds_addr, global_addr, type},
// group1 = {data_size, tensor dims/strides, tile dims}. Tracked by TENSORcnt.
__device__ __forceinline__
void tdm_load_tile_f16(unsigned lds_byte_off, const void* gaddr,
                       unsigned tensor_d0, unsigned tensor_d1,
                       unsigned stride_d0_elems)
{
    unsigned long long ga = (unsigned long long)gaddr;
    u32x4 g0;
    g0[0] = 1u;                                           // count=1, user descriptor
    g0[1] = lds_byte_off;                                 // lds_addr (bytes)
    g0[2] = (unsigned)(ga & 0xffffffffu);                 // global_addr[31:0]
    g0[3] = (unsigned)((ga >> 32) & 0x1ffffffu)           // global_addr[56:32]
          | (2u << 30);                                   // type = 2 ("image")
    i32x8 g1;
    g1[0] = 0x10000;                                      // wg_mask=0, data_size=1 (2B)
    g1[1] = (int)((tensor_d0 & 0xffffu) << 16);           // tensor_dim0[15:0]
    g1[2] = (int)((tensor_d0 >> 16) |
                  ((tensor_d1 & 0xffffu) << 16));         // dim0[31:16] | dim1[15:0]
    g1[3] = (int)((tensor_d1 >> 16) | ((unsigned)BK << 16)); // dim1[31:16] | tile_dim0=32
    g1[4] = BM;                                           // tile_dim1=128, tile_dim2=0
    g1[5] = (int)stride_d0_elems;                         // tensor_dim0_stride[31:0]
    g1[6] = 0;                                            // stride0[47:32], dim1_stride lo
    g1[7] = 0;
    i32x4 g2 = {1, 0, 0, 0};                              // tensor_dim2=1, rest unused
    i32x4 g3 = {0, 0, 0, 0};
#if defined(__clang_major__) && (__clang_major__ >= 23)
    i32x8 g4 = {0, 0, 0, 0, 0, 0, 0, 0};
    __builtin_amdgcn_tensor_load_to_lds(g0, g1, g2, g3, g4, 0);
#else
    __builtin_amdgcn_tensor_load_to_lds(g0, g1, g2, g3, 0);
#endif
}

// ---------------- main WMMA GEMM ----------------
// out[m,n] = (sum_k A[m,k]*W[n,k]) * scale[n] + bias[n]
__global__ __launch_bounds__(THREADS)
void fp4_linear_wmma(const _Float16* __restrict__ A,
                     const _Float16* __restrict__ W,
                     const float* __restrict__ scale,
                     const float* __restrict__ bias,
                     float* __restrict__ out,
                     int M, int N, int K)
{
    alignas(16) __shared__ _Float16 As[2][BM * BK];  // 8 KiB x2
    alignas(16) __shared__ _Float16 Ws[2][BN * BK];  // 8 KiB x2

    const int tid   = threadIdx.x;
    const int lane  = tid & 31;
    const int wave  = tid >> 5;      // 0..7
    const int wm    = wave >> 2;     // 0..1 (M direction, 64 rows)
    const int wn    = wave & 3;      // 0..3 (N direction, 32 cols)
    const int lrow  = lane & 15;
    const int lhalf = lane >> 4;

    const int m0 = blockIdx.y * BM;
    const int n0 = blockIdx.x * BN;

    // LDS byte offsets of the four tile buffers (generic addr low 32 bits == LDS offset)
    const unsigned ldsA[2] = { (unsigned)(unsigned long long)(void*)&As[0][0],
                               (unsigned)(unsigned long long)(void*)&As[1][0] };
    const unsigned ldsW[2] = { (unsigned)(unsigned long long)(void*)&Ws[0][0],
                               (unsigned)(unsigned long long)(void*)&Ws[1][0] };

    const _Float16* gA = A + (size_t)m0 * K;   // tile start advances along K
    const _Float16* gW = W + (size_t)n0 * K;

    v8f acc[4][2] = {};
    const int nk = K / BK;

    // prologue: TDM-load tile 0 into buffer 0 (wave 0 drives the DMA)
    if (wave == 0) {
        tdm_load_tile_f16(ldsA[0], gA, (unsigned)K, (unsigned)BM, (unsigned)K);
        tdm_load_tile_f16(ldsW[0], gW, (unsigned)K, (unsigned)BN, (unsigned)K);
    }

    for (int kt = 0; kt < nk; ++kt) {
        const int cur = kt & 1;
        const int nxt = cur ^ 1;

        if (wave == 0) {
            if (kt + 1 < nk) {
                // kick off DMA of the next tile, then wait for the current pair:
                // TENSORcnt completes in order, so <=2 means the older 2 landed.
                tdm_load_tile_f16(ldsA[nxt], gA + (size_t)(kt + 1) * BK,
                                  (unsigned)K, (unsigned)BM, (unsigned)K);
                tdm_load_tile_f16(ldsW[nxt], gW + (size_t)(kt + 1) * BK,
                                  (unsigned)K, (unsigned)BN, (unsigned)K);
                __builtin_amdgcn_s_wait_tensorcnt(2);
            } else {
                __builtin_amdgcn_s_wait_tensorcnt(0);
            }
        }
        __syncthreads();   // current buffer visible to all waves

        // A fragments per ISA 16-bit A layout:
        // lane half h: K in [8h, 8h+8) (v0-3) and [16+8h, 16+8h+8) (v4-7)
        v16h afrag[4];
#pragma unroll
        for (int mi = 0; mi < 4; ++mi) {
            const int mr = wm * 64 + mi * 16 + lrow;
            v8h lo = *(const v8h*)&As[cur][mr * BK + 8 * lhalf];
            v8h hi = *(const v8h*)&As[cur][mr * BK + 16 + 8 * lhalf];
            afrag[mi] = __builtin_shufflevector(lo, hi,
                0, 1, 2, 3, 4, 5, 6, 7, 8, 9, 10, 11, 12, 13, 14, 15);
        }

#pragma unroll
        for (int ni = 0; ni < 2; ++ni) {
            const int nr = wn * 32 + ni * 16 + lrow;   // B mirrors A (W is [N,K])
            v8h lo = *(const v8h*)&Ws[cur][nr * BK + 8 * lhalf];
            v8h hi = *(const v8h*)&Ws[cur][nr * BK + 16 + 8 * lhalf];
            v16h bfrag = __builtin_shufflevector(lo, hi,
                0, 1, 2, 3, 4, 5, 6, 7, 8, 9, 10, 11, 12, 13, 14, 15);
#pragma unroll
            for (int mi = 0; mi < 4; ++mi) {
                acc[mi][ni] = __builtin_amdgcn_wmma_f32_16x16x32_f16(
                    false, afrag[mi], false, bfrag,
                    (short)0, acc[mi][ni], false, false);
            }
        }

        __syncthreads();   // everyone done reading `cur` before TDM reuses it
    }

    // epilogue: C/D layout — VGPR r holds (M = r + 8*lhalf, N = lrow)
#pragma unroll
    for (int ni = 0; ni < 2; ++ni) {
        const int col = n0 + wn * 32 + ni * 16 + lrow;
        const float sc = scale[col];
        const float bs = bias[col];
#pragma unroll
        for (int mi = 0; mi < 4; ++mi) {
            const int mbase = m0 + wm * 64 + mi * 16 + 8 * lhalf;
#pragma unroll
            for (int r = 0; r < 8; ++r) {
                out[(size_t)(mbase + r) * N + col] = acc[mi][ni][r] * sc + bs;
            }
        }
    }
}

// ---------------- launcher ----------------
extern "C" void kernel_launch(void* const* d_in, const int* in_sizes, int n_in,
                              void* d_out, int out_size, void* d_ws, size_t ws_size,
                              hipStream_t stream) {
    const float* x    = (const float*)d_in[0];
    const int*   wq   = (const int*)d_in[1];
    const float* wsc  = (const float*)d_in[2];
    const float* bias = (const float*)d_in[3];
    float*       out  = (float*)d_out;

    const int OUT = in_sizes[2];              // 4096
    const int IN  = in_sizes[1] / OUT;        // 4096 (= K)
    const int Mg  = in_sizes[0] / IN;         // 8192 (= B*S)

    _Float16* Ah = (_Float16*)d_ws;                    // [M,K] f16: 64 MiB
    _Float16* Wh = Ah + (size_t)Mg * IN;               // [N,K] f16: 32 MiB

    cvt_f32_to_f16<<<in_sizes[0] / (256 * 8), 256, 0, stream>>>(x, Ah);
    cvt_i32_to_f16<<<in_sizes[1] / (256 * 8), 256, 0, stream>>>(wq, Wh);

    dim3 grid(OUT / BN, Mg / BM);             // 32 x 64 = 2048 workgroups
    fp4_linear_wmma<<<grid, THREADS, 0, stream>>>(Ah, Wh, wsc, bias, out,
                                                  Mg, OUT, IN);
}
